// ExecutionTemporalGNN_23673859736030
// MI455X (gfx1250) — matrix operand order repaired and checked by reading, not compiled, not run on hardware
//
#include <hip/hip_runtime.h>
#include <hip/hip_bf16.h>

// ---------------------------------------------------------------------------
// ExecutionTemporalGNN on MI455X (gfx1250, wave32).
// All dense GEMMs use v_wmma_f32_16x16x32_f16 (f16 inputs, f32 accum);
// each wave computes a 16x64 C strip (4 WMMA column tiles) so the A fragment
// is loaded once per k-step and reused 4x. The column-tile count is resolved
// once per wave and dispatched to a templated branch-free inner loop so the
// compiler can software-pipeline loads against WMMAs.
// Irregular graph ops (segment softmax / scatter) use global atomics.
// ---------------------------------------------------------------------------

#define NNODES 10000
#define TSTEPS 6
#define FT 8
#define SDIM 64
#define EDIM 4
#define HDIM 128
#define NHEADS 4
#define DHEAD 32
#define NEDGES 160000
#define ENF (NEDGES + NNODES)   // edges + self loops = 170000
#define KCOMB 96                // 72 padded to multiple of 32
#define NEG_SLOPE 0.2f

typedef __attribute__((ext_vector_type(16))) _Float16 v16h;
typedef __attribute__((ext_vector_type(8)))  _Float16 v8h;
typedef __attribute__((ext_vector_type(8)))  float    v8f;

// ------------------------------ ws layout ----------------------------------
static constexpr size_t AL2(size_t x) { return (x + 255) & ~(size_t)255; }
// persistent
static constexpr size_t OFF_SRCF  = 0;
static constexpr size_t OFF_DSTF  = AL2(OFF_SRCF + (size_t)ENF * 4);
static constexpr size_t OFF_SEQ16 = AL2(OFF_DSTF + (size_t)ENF * 4);
static constexpr size_t OFF_WFE   = AL2(OFF_SEQ16 + (size_t)TSTEPS * NNODES * HDIM * 2);
static constexpr size_t OFF_WGL0  = AL2(OFF_WFE  + (size_t)HDIM * KCOMB * 2);
static constexpr size_t OFF_WGR0  = AL2(OFF_WGL0 + (size_t)HDIM * HDIM * 2);
static constexpr size_t OFF_WGE0  = AL2(OFF_WGR0 + (size_t)HDIM * HDIM * 2);
static constexpr size_t OFF_WGL1  = AL2(OFF_WGE0 + (size_t)HDIM * HDIM * 2);
static constexpr size_t OFF_WGR1  = AL2(OFF_WGL1 + (size_t)HDIM * HDIM * 2);
static constexpr size_t OFF_WGE1  = AL2(OFF_WGR1 + (size_t)HDIM * HDIM * 2);
static constexpr size_t OFF_WIH0  = AL2(OFF_WGE1 + (size_t)HDIM * HDIM * 2);
static constexpr size_t OFF_WHH0  = AL2(OFF_WIH0 + (size_t)3 * HDIM * HDIM * 2);
static constexpr size_t OFF_WIH1  = AL2(OFF_WHH0 + (size_t)3 * HDIM * HDIM * 2);
static constexpr size_t OFF_WHH1  = AL2(OFF_WIH1 + (size_t)3 * HDIM * HDIM * 2);
static constexpr size_t OFF_WMIN  = AL2(OFF_WHH1 + (size_t)3 * HDIM * HDIM * 2);
static constexpr size_t OFF_WMOUT = AL2(OFF_WMIN + (size_t)3 * HDIM * HDIM * 2);
static constexpr size_t OFF_WOH1  = AL2(OFF_WMOUT + (size_t)HDIM * HDIM * 2);
static constexpr size_t OFF_WDH1  = AL2(OFF_WOH1 + (size_t)HDIM * HDIM * 2);
static constexpr size_t OFF_WEH1  = AL2(OFF_WDH1 + (size_t)HDIM * HDIM * 2);
static constexpr size_t OFF_WPH1  = AL2(OFF_WEH1 + (size_t)64 * HDIM * 2);
static constexpr size_t OFF_WCH1  = AL2(OFF_WPH1 + (size_t)HDIM * HDIM * 2);
static constexpr size_t OFF_WOH2  = AL2(OFF_WCH1 + (size_t)32 * HDIM * 2);
static constexpr size_t OFF_WDH2  = AL2(OFF_WOH2 + (size_t)1 * HDIM * 2);
static constexpr size_t OFF_WEH2  = AL2(OFF_WDH2 + (size_t)4 * HDIM * 2);
static constexpr size_t OFF_WPH2  = AL2(OFF_WEH2 + (size_t)3 * 64 * 2);
static constexpr size_t OFF_WCH2  = AL2(OFF_WPH2 + (size_t)4 * HDIM * 2);
static constexpr size_t OFF_ARENA = AL2(OFF_WCH2 + (size_t)1 * 32 * 2);
// phase A (graph + GAT timestep loop)
static constexpr size_t A_EATTR  = OFF_ARENA;
static constexpr size_t A_EPROJ0 = AL2(A_EATTR  + (size_t)ENF * HDIM * 2);
static constexpr size_t A_EPROJ1 = AL2(A_EPROJ0 + (size_t)ENF * HDIM * 2);
static constexpr size_t A_XL     = AL2(A_EPROJ1 + (size_t)ENF * HDIM * 2);
static constexpr size_t A_XR     = AL2(A_XL     + (size_t)NNODES * HDIM * 4);
static constexpr size_t A_FEOUT  = AL2(A_XR     + (size_t)NNODES * HDIM * 4);
static constexpr size_t A_LOGITS = AL2(A_FEOUT  + (size_t)NNODES * HDIM * 4);
static constexpr size_t A_SEGMAX = AL2(A_LOGITS + (size_t)ENF * 4 * 4);
static constexpr size_t A_SEGDEN = AL2(A_SEGMAX + (size_t)NNODES * 4 * 4);
static constexpr size_t A_GATOUT = AL2(A_SEGDEN + (size_t)NNODES * 4 * 4);
static constexpr size_t A_LOOPS  = AL2(A_GATOUT + (size_t)NNODES * HDIM * 4);
static constexpr size_t A_CNT    = AL2(A_LOOPS  + (size_t)NNODES * HDIM * 4);
static constexpr size_t A_XCOMB  = AL2(A_CNT    + (size_t)NNODES * 4);
static constexpr size_t A_H32    = AL2(A_XCOMB  + (size_t)NNODES * KCOMB * 2);
static constexpr size_t A_H16    = AL2(A_H32    + (size_t)NNODES * HDIM * 4);
static constexpr size_t A_GNM    = AL2(A_H16    + (size_t)NNODES * HDIM * 2);
static constexpr size_t A_GNV    = AL2(A_GNM    + (size_t)HDIM * 4);
// phase B (GRU / MHA / heads) aliases the phase-A arena
static constexpr size_t B_GI     = OFF_ARENA;
static constexpr size_t B_GH     = AL2(B_GI     + (size_t)NNODES * 3 * HDIM * 4);
static constexpr size_t B_GRUH32 = AL2(B_GH     + (size_t)NNODES * 3 * HDIM * 4);
static constexpr size_t B_GRUH16 = AL2(B_GRUH32 + (size_t)NNODES * HDIM * 4);
static constexpr size_t B_YS1    = AL2(B_GRUH16 + (size_t)NNODES * HDIM * 2);
static constexpr size_t B_HGRU16 = AL2(B_YS1    + (size_t)TSTEPS * NNODES * HDIM * 2);
static constexpr size_t B_HLAST  = AL2(B_HGRU16 + (size_t)TSTEPS * NNODES * HDIM * 2);
static constexpr size_t B_QKV16  = AL2(B_HLAST  + (size_t)NNODES * HDIM * 4);
static constexpr size_t B_AW     = AL2(B_QKV16  + (size_t)TSTEPS * NNODES * 3 * HDIM * 2);
static constexpr size_t B_ATTN16 = AL2(B_AW     + (size_t)NNODES * NHEADS * 36 * 4);
static constexpr size_t B_HATTN  = AL2(B_ATTN16 + (size_t)NNODES * HDIM * 2);
static constexpr size_t B_HFIN32 = AL2(B_HATTN  + (size_t)NNODES * HDIM * 4);
static constexpr size_t B_HFIN16 = AL2(B_HFIN32 + (size_t)NNODES * HDIM * 4);
static constexpr size_t B_MID16  = AL2(B_HFIN16 + (size_t)NNODES * HDIM * 2);
static constexpr size_t B_EXCT   = AL2(B_MID16  + (size_t)NNODES * HDIM * 2);

// output element offsets (f32 out, 490000 total)
static constexpr size_t O_ORDER = 0;
static constexpr size_t O_DEM   = 10000;
static constexpr size_t O_EXC   = 50000;
static constexpr size_t O_PROP  = 80000;
static constexpr size_t O_CONF  = 120000;
static constexpr size_t O_AW    = 130000;

// ------------------------------ device helpers -----------------------------
__device__ __forceinline__ float sigmf(float x) { return 1.f / (1.f + expf(-x)); }
__device__ __forceinline__ unsigned encf(float f) {
  unsigned u = __float_as_uint(f);
  return (u & 0x80000000u) ? ~u : (u ^ 0x80000000u);
}
__device__ __forceinline__ float decf(unsigned e) {
  unsigned u = (e & 0x80000000u) ? (e ^ 0x80000000u) : ~e;
  return __uint_as_float(u);
}

// ------------------------------ utility kernels ----------------------------
__global__ void k_zero_f32(float* p, long long n) {
  long long i = (long long)blockIdx.x * blockDim.x + threadIdx.x;
  if (i < n) p[i] = 0.f;
}
__global__ void k_zero_f16(_Float16* p, long long n) {
  long long i = (long long)blockIdx.x * blockDim.x + threadIdx.x;
  if (i < n) p[i] = (_Float16)0.f;
}
__global__ void k_fill_u32(unsigned* p, unsigned v, long long n) {
  long long i = (long long)blockIdx.x * blockDim.x + threadIdx.x;
  if (i < n) p[i] = v;
}
// dst[r*cd+c] = c<cs ? src[r*cs+c] : 0   (f32 -> f16, optional K pad)
__global__ void k_convert_f16(const float* __restrict__ src, _Float16* __restrict__ dst,
                              int rows, int cs, int cd) {
  long long i = (long long)blockIdx.x * blockDim.x + threadIdx.x;
  if (i >= (long long)rows * cd) return;
  int r = (int)(i / cd), c = (int)(i % cd);
  dst[i] = (c < cs) ? (_Float16)src[(size_t)r * cs + c] : (_Float16)0.f;
}
// src[Ks,Ns] f32 -> dst[Ns,Kd] f16 (transposed, K padded with zeros)
__global__ void k_transpose_f16(const float* __restrict__ src, _Float16* __restrict__ dst,
                                int Ks, int Ns, int Kd) {
  long long i = (long long)blockIdx.x * blockDim.x + threadIdx.x;
  if (i >= (long long)Ns * Kd) return;
  int n = (int)(i / Kd), k = (int)(i % Kd);
  dst[i] = (k < Ks) ? (_Float16)src[(size_t)k * Ns + n] : (_Float16)0.f;
}

// ------------------------------ WMMA GEMM ----------------------------------
// C[M,Nn] = act(A[M,K]_f16 * BT[Nn,K]_f16^T + bias), K % 32 == 0.
// One wave owns a 16x(16*GNT) C strip: A fragment loaded once per k-step and
// reused across the strip's column tiles. Tile count resolved once per wave,
// dispatched to a branch-free templated k-loop so loads pipeline with WMMAs.
// act: 0 none, 1 relu, 2 sigmoid. C (f32) and/or Ch (f16) may be null.
#define GNT 4

template <int JMAX>
__device__ __forceinline__ void gemm_strip(const _Float16* __restrict__ ap,
                                           const _Float16* const* bp, int K, v8f* acc) {
  for (int k0 = 0; k0 < K; k0 += 32) {
    v8h alo = *(const v8h*)(ap + k0);
    v8h ahi = *(const v8h*)(ap + k0 + 8);
    v16h av = __builtin_shufflevector(alo, ahi, 0,1,2,3,4,5,6,7,8,9,10,11,12,13,14,15);
#pragma unroll
    for (int j = 0; j < JMAX; ++j) {
      v16h bvec = *(const v16h*)(bp[j] + k0);
      acc[j] = __builtin_amdgcn_wmma_f32_16x16x32_f16(false, av, false, bvec,
                                                      (short)0, acc[j], false, false);
    }
  }
}

__global__ __launch_bounds__(128) void k_gemm_wmma(
    const _Float16* __restrict__ A, const _Float16* __restrict__ BT,
    const float* __restrict__ bias, float* __restrict__ C, _Float16* __restrict__ Ch,
    int M, int Nn, int K, int act) {
  int wave = threadIdx.x >> 5;
  int lane = threadIdx.x & 31;
  int tm = blockIdx.x * 4 + wave;   // 16-row tile
  int tn0 = blockIdx.y * GNT;       // first 16-col tile of this wave's strip
  if (tm * 16 >= M) return;         // uniform per wave
  int ntiles = (Nn + 15) >> 4;
  int jmax = ntiles - tn0;
  jmax = jmax > GNT ? GNT : jmax;   // wave-uniform column-tile count
  int r0 = tm * 16 + (lane & 15);
  int arow = (r0 < M) ? r0 : (M - 1);
  // A fragment: lane m=lane&15; 16 contiguous K halves at (lane>>4)*16
  const _Float16* ap = A + (size_t)arow * K + ((lane >> 4) << 4);
  // B fragment: lane n=lane&15; 16 contiguous K halves at (lane>>4)*16
  const _Float16* bp[GNT];
#pragma unroll
  for (int j = 0; j < GNT; ++j) {
    int c0 = (tn0 + j) * 16 + (lane & 15);
    int bcol = (c0 < Nn) ? c0 : (Nn - 1);
    bp[j] = BT + (size_t)bcol * K + ((lane >> 4) << 4);
  }
  v8f acc[GNT];
#pragma unroll
  for (int j = 0; j < GNT; ++j) acc[j] = (v8f){0.f, 0.f, 0.f, 0.f, 0.f, 0.f, 0.f, 0.f};
  switch (jmax) {
    case 4: gemm_strip<4>(ap, bp, K, acc); break;
    case 3: gemm_strip<3>(ap, bp, K, acc); break;
    case 2: gemm_strip<2>(ap, bp, K, acc); break;
    default: gemm_strip<1>(ap, bp, K, acc); break;
  }
  int mbase = tm * 16 + ((lane >> 4) << 3);
  for (int j = 0; j < jmax; ++j) {
    int nI = (tn0 + j) * 16 + (lane & 15);
    if (nI >= Nn) continue;
    float bb = bias ? bias[nI] : 0.f;
#pragma unroll
    for (int r = 0; r < 8; ++r) {
      int m = mbase + r;
      if (m >= M) continue;
      float v = acc[j][r] + bb;
      if (act == 1) v = v > 0.f ? v : 0.f;
      else if (act == 2) v = sigmf(v);
      if (C)  C[(size_t)m * Nn + nI] = v;
      if (Ch) Ch[(size_t)m * Nn + nI] = (_Float16)v;
    }
  }
}

// ------------------------------ graph kernels ------------------------------
__global__ void k_count(const int* __restrict__ dst, float* __restrict__ cnt, int E) {
  int e = blockIdx.x * blockDim.x + threadIdx.x;
  if (e < E) atomicAdd(&cnt[dst[e]], 1.f);
}
// edge encoder: relu(edge_attr @ ee_w + ee_b); write f16 + accumulate loop sum
__global__ void k_edge_enc(const float* __restrict__ eattr, const int* __restrict__ dst,
                           const float* __restrict__ w, const float* __restrict__ b,
                           _Float16* __restrict__ eattr16, float* __restrict__ loopsum) {
  long long i = (long long)blockIdx.x * blockDim.x + threadIdx.x;
  if (i >= (long long)NEDGES * HDIM) return;
  int e = (int)(i >> 7), h = (int)(i & 127);
  float v = b[h];
#pragma unroll
  for (int j = 0; j < EDIM; ++j) v += eattr[(size_t)e * EDIM + j] * w[j * HDIM + h];
  v = v > 0.f ? v : 0.f;
  eattr16[i] = (_Float16)v;
  atomicAdd(&loopsum[(size_t)dst[e] * HDIM + h], v);
}
__global__ void k_loopfill(const float* __restrict__ loopsum, const float* __restrict__ cnt,
                           _Float16* __restrict__ eattr16) {
  long long i = (long long)blockIdx.x * blockDim.x + threadIdx.x;
  if (i >= (long long)NNODES * HDIM) return;
  int n = (int)(i >> 7);
  eattr16[(size_t)(NEDGES + n) * HDIM + (i & 127)] = (_Float16)(loopsum[i] / fmaxf(cnt[n], 1.f));
}
__global__ void k_fill_idx(const int* __restrict__ src, const int* __restrict__ dst,
                           int* __restrict__ srcf, int* __restrict__ dstf) {
  int i = blockIdx.x * blockDim.x + threadIdx.x;
  if (i >= ENF) return;
  srcf[i] = (i < NEDGES) ? src[i] : (i - NEDGES);
  dstf[i] = (i < NEDGES) ? dst[i] : (i - NEDGES);
}
__global__ void k_xcomb(const float* __restrict__ xt, const float* __restrict__ se,
                        _Float16* __restrict__ xc, int t) {
  long long i = (long long)blockIdx.x * blockDim.x + threadIdx.x;
  if (i >= (long long)NNODES * KCOMB) return;
  int n = (int)(i / KCOMB), c = (int)(i % KCOMB);
  float v = 0.f;
  if (c < FT) v = xt[((size_t)t * NNODES + n) * FT + c];
  else if (c < FT + SDIM) v = se[(size_t)n * SDIM + (c - FT)];
  xc[i] = (_Float16)v;
}
// per-row LayerNorm(+relu), C==128, one block per row
__global__ __launch_bounds__(128) void k_ln_relu(const float* __restrict__ x,
    const float* __restrict__ g, const float* __restrict__ b,
    float* __restrict__ h32, _Float16* __restrict__ h16, int relu) {
  __shared__ float sb[HDIM];
  int row = blockIdx.x, c = threadIdx.x;
  float v = x[(size_t)row * HDIM + c];
  sb[c] = v; __syncthreads();
  for (int s = 64; s > 0; s >>= 1) { if (c < s) sb[c] += sb[c + s]; __syncthreads(); }
  float mean = sb[0] / HDIM; __syncthreads();
  float d = v - mean;
  sb[c] = d * d; __syncthreads();
  for (int s = 64; s > 0; s >>= 1) { if (c < s) sb[c] += sb[c + s]; __syncthreads(); }
  float var = sb[0] / HDIM;
  float y = g[c] * d * rsqrtf(var + 1e-5f) + b[c];
  if (relu) y = y > 0.f ? y : 0.f;
  h32[(size_t)row * HDIM + c] = y;
  h16[(size_t)row * HDIM + c] = (_Float16)y;
}
// LN(a+b), no relu
__global__ __launch_bounds__(128) void k_addln(const float* __restrict__ a,
    const float* __restrict__ bb, const float* __restrict__ g, const float* __restrict__ be,
    float* __restrict__ h32, _Float16* __restrict__ h16) {
  __shared__ float sb[HDIM];
  int row = blockIdx.x, c = threadIdx.x;
  float v = a[(size_t)row * HDIM + c] + bb[(size_t)row * HDIM + c];
  sb[c] = v; __syncthreads();
  for (int s = 64; s > 0; s >>= 1) { if (c < s) sb[c] += sb[c + s]; __syncthreads(); }
  float mean = sb[0] / HDIM; __syncthreads();
  float d = v - mean;
  sb[c] = d * d; __syncthreads();
  for (int s = 64; s > 0; s >>= 1) { if (c < s) sb[c] += sb[c + s]; __syncthreads(); }
  float var = sb[0] / HDIM;
  float y = g[c] * d * rsqrtf(var + 1e-5f) + be[c];
  h32[(size_t)row * HDIM + c] = y;
  h16[(size_t)row * HDIM + c] = (_Float16)y;
}
// GATv2 logits: s = leaky(xl[src]+xr[dst]+eproj); logit = <s, att[h]>; seg max
__global__ void k_gat_logits(const float* __restrict__ xl, const float* __restrict__ xr,
    const _Float16* __restrict__ eproj, const int* __restrict__ srcf,
    const int* __restrict__ dstf, const float* __restrict__ att,
    float* __restrict__ logits, unsigned* __restrict__ segmax) {
  long long i = (long long)blockIdx.x * blockDim.x + threadIdx.x;
  if (i >= (long long)ENF * NHEADS) return;
  int ef = (int)(i >> 2), h = (int)(i & 3);
  int s = srcf[ef], d = dstf[ef];
  const float* xls = xl + (size_t)s * HDIM + h * DHEAD;
  const float* xrd = xr + (size_t)d * HDIM + h * DHEAD;
  const _Float16* ep = eproj + (size_t)ef * HDIM + h * DHEAD;
  const float* at = att + h * DHEAD;
  float lg = 0.f;
#pragma unroll 8
  for (int k = 0; k < DHEAD; ++k) {
    float t = xls[k] + xrd[k] + (float)ep[k];
    t = t > 0.f ? t : NEG_SLOPE * t;
    lg += t * at[k];
  }
  logits[i] = lg;
  atomicMax(&segmax[d * NHEADS + h], encf(lg));
}
__global__ void k_gat_exp(float* __restrict__ logits, const unsigned* __restrict__ segmax,
                          float* __restrict__ segden, const int* __restrict__ dstf) {
  long long i = (long long)blockIdx.x * blockDim.x + threadIdx.x;
  if (i >= (long long)ENF * NHEADS) return;
  int ef = (int)(i >> 2), h = (int)(i & 3);
  int d = dstf[ef];
  float a = expf(logits[i] - decf(segmax[d * NHEADS + h]));
  logits[i] = a;
  atomicAdd(&segden[d * NHEADS + h], a);
}
__global__ void k_bias_init(float* __restrict__ out, const float* __restrict__ b) {
  long long i = (long long)blockIdx.x * blockDim.x + threadIdx.x;
  if (i >= (long long)NNODES * HDIM) return;
  out[i] = b[i & 127];
}
__global__ void k_gat_scatter(const float* __restrict__ xl, const float* __restrict__ logits,
    const float* __restrict__ segden, const int* __restrict__ srcf,
    const int* __restrict__ dstf, float* __restrict__ out) {
  long long i = (long long)blockIdx.x * blockDim.x + threadIdx.x;
  if (i >= (long long)ENF * HDIM) return;
  int ef = (int)(i >> 7), hd = (int)(i & 127), h = hd >> 5;
  int s = srcf[ef], d = dstf[ef];
  float alpha = logits[(size_t)ef * NHEADS + h] / segden[(size_t)d * NHEADS + h];
  atomicAdd(&out[(size_t)d * HDIM + hd], xl[(size_t)s * HDIM + hd] * alpha);
}
// GraphNorm column reductions (one block per feature)
__global__ __launch_bounds__(256) void k_colmean(const float* __restrict__ x,
                                                 float* __restrict__ mean) {
  __shared__ float sb[256];
  int c = blockIdx.x, tid = threadIdx.x;
  float s = 0.f;
  for (int n = tid; n < NNODES; n += 256) s += x[(size_t)n * HDIM + c];
  sb[tid] = s; __syncthreads();
  for (int k = 128; k > 0; k >>= 1) { if (tid < k) sb[tid] += sb[tid + k]; __syncthreads(); }
  if (tid == 0) mean[c] = sb[0] / NNODES;
}
__global__ __launch_bounds__(256) void k_colvar(const float* __restrict__ x,
    const float* __restrict__ mean, const float* __restrict__ ms, float* __restrict__ var) {
  __shared__ float sb[256];
  int c = blockIdx.x, tid = threadIdx.x;
  float mm = ms[c] * mean[c], s = 0.f;
  for (int n = tid; n < NNODES; n += 256) {
    float t = x[(size_t)n * HDIM + c] - mm; s += t * t;
  }
  sb[tid] = s; __syncthreads();
  for (int k = 128; k > 0; k >>= 1) { if (tid < k) sb[tid] += sb[tid + k]; __syncthreads(); }
  if (tid == 0) var[c] = sb[0] / NNODES;
}
// h += relu(GraphNorm(x)); update f32 + f16 copies
__global__ void k_gn_apply(const float* __restrict__ x, const float* __restrict__ mean,
    const float* __restrict__ var, const float* __restrict__ ms,
    const float* __restrict__ w, const float* __restrict__ b,
    float* __restrict__ h32, _Float16* __restrict__ h16) {
  long long i = (long long)blockIdx.x * blockDim.x + threadIdx.x;
  if (i >= (long long)NNODES * HDIM) return;
  int c = (int)(i & 127);
  float t = x[i] - ms[c] * mean[c];
  float y = w[c] * t * rsqrtf(var[c] + 1e-5f) + b[c];
  y = y > 0.f ? y : 0.f;
  float hn = h32[i] + y;
  h32[i] = hn;
  h16[i] = (_Float16)hn;
}
// GRU pointwise
__global__ void k_gru_pw(const float* __restrict__ gi, const float* __restrict__ gh,
    float* __restrict__ h32, _Float16* __restrict__ h16, _Float16* __restrict__ ys16,
    float* __restrict__ ylast, int t, int interleave) {
  long long i = (long long)blockIdx.x * blockDim.x + threadIdx.x;
  if (i >= (long long)NNODES * HDIM) return;
  int n = (int)(i >> 7), c = (int)(i & 127);
  size_t gb = (size_t)n * 3 * HDIM + c;
  float ir = gi[gb], iz = gi[gb + HDIM], inn = gi[gb + 2 * HDIM];
  float hr = gh[gb], hz = gh[gb + HDIM], hc = gh[gb + 2 * HDIM];
  float hprev = h32[i];
  float r = sigmf(ir + hr), z = sigmf(iz + hz);
  float nc = tanhf(inn + r * hc);
  float hnew = (1.f - z) * nc + z * hprev;
  h32[i] = hnew;
  h16[i] = (_Float16)hnew;
  size_t yi = interleave ? (((size_t)n * TSTEPS + t) * HDIM + c) : (size_t)i;
  ys16[yi] = (_Float16)hnew;
  if (ylast) ylast[i] = hnew;
}
// MHA over T=6 per (node, head): full aw, output only t = T-1
__global__ void k_attention(const _Float16* __restrict__ qkv, float* __restrict__ aw,
                            _Float16* __restrict__ attn_last) {
  int i = blockIdx.x * blockDim.x + threadIdx.x;
  if (i >= NNODES * NHEADS) return;
  int n = i >> 2, h = i & 3;
  const _Float16* base = qkv + (size_t)n * TSTEPS * (3 * HDIM);
  float sc[TSTEPS][TSTEPS];
  const float scale = 0.17677669529663687f;  // 1/sqrt(32)
  for (int t = 0; t < TSTEPS; ++t)
    for (int s = 0; s < TSTEPS; ++s) {
      float acc = 0.f;
      const _Float16* q = base + (size_t)t * 3 * HDIM + h * DHEAD;
      const _Float16* kk = base + (size_t)s * 3 * HDIM + HDIM + h * DHEAD;
#pragma unroll 8
      for (int k = 0; k < DHEAD; ++k) acc += (float)q[k] * (float)kk[k];
      sc[t][s] = acc * scale;
    }
  float* awp = aw + (size_t)i * 36;
  for (int t = 0; t < TSTEPS; ++t) {
    float m = sc[t][0];
    for (int s = 1; s < TSTEPS; ++s) m = fmaxf(m, sc[t][s]);
    float den = 0.f;
    for (int s = 0; s < TSTEPS; ++s) { sc[t][s] = expf(sc[t][s] - m); den += sc[t][s]; }
    float inv = 1.f / den;
    for (int s = 0; s < TSTEPS; ++s) { sc[t][s] *= inv; awp[t * TSTEPS + s] = sc[t][s]; }
  }
  // output for last timestep only
  for (int k = 0; k < DHEAD; ++k) {
    float o = 0.f;
    for (int s = 0; s < TSTEPS; ++s)
      o += sc[TSTEPS - 1][s] * (float)base[(size_t)s * 3 * HDIM + 2 * HDIM + h * DHEAD + k];
    attn_last[(size_t)n * HDIM + h * DHEAD + k] = (_Float16)o;
  }
}
__global__ void k_awmean(const float* __restrict__ aw, float* __restrict__ out_aw) {
  long long i = (long long)blockIdx.x * blockDim.x + threadIdx.x;
  if (i >= (long long)NNODES * 36) return;
  int n = (int)(i / 36), ts = (int)(i % 36);
  float s = 0.f;
  for (int h = 0; h < NHEADS; ++h) s += aw[((size_t)n * NHEADS + h) * 36 + ts];
  out_aw[i] = s * 0.25f;
}
__global__ void k_softmax3(const float* __restrict__ x, float* __restrict__ out) {
  int n = blockIdx.x * blockDim.x + threadIdx.x;
  if (n >= NNODES) return;
  float a = x[n * 3], b = x[n * 3 + 1], c = x[n * 3 + 2];
  float m = fmaxf(a, fmaxf(b, c));
  float ea = expf(a - m), eb = expf(b - m), ec = expf(c - m);
  float inv = 1.f / (ea + eb + ec);
  out[n * 3] = ea * inv; out[n * 3 + 1] = eb * inv; out[n * 3 + 2] = ec * inv;
}

// ------------------------------ host side ----------------------------------
extern "C" void kernel_launch(void* const* d_in, const int* in_sizes, int n_in,
                              void* d_out, int out_size, void* d_ws, size_t ws_size,
                              hipStream_t stream) {
  (void)in_sizes; (void)n_in; (void)out_size; (void)ws_size;
  // inputs (setup_inputs dict order, params in _make_params insertion order)
  const float* x_temporal = (const float*)d_in[0];
  const float* s_emb      = (const float*)d_in[1];
  const int*   eidx       = (const int*)d_in[2];
  const float* e_attr     = (const float*)d_in[3];
  const float* fe_w  = (const float*)d_in[4];  const float* fe_b  = (const float*)d_in[5];
  const float* fe_lg = (const float*)d_in[6];  const float* fe_lb = (const float*)d_in[7];
  const float* ee_w  = (const float*)d_in[8];  const float* ee_b  = (const float*)d_in[9];
  const float *gat_wl[2], *gat_wr[2], *gat_we[2], *gat_att[2], *gat_b[2];
  const float *gn_w[2], *gn_b[2], *gn_ms[2];
  const float *gru_wih[2], *gru_whh[2], *gru_bih[2], *gru_bhh[2];
  for (int l = 0; l < 2; ++l) {
    int base = 10 + l * 12;
    gat_wl[l]  = (const float*)d_in[base + 0];
    gat_wr[l]  = (const float*)d_in[base + 1];
    gat_we[l]  = (const float*)d_in[base + 2];
    gat_att[l] = (const float*)d_in[base + 3];
    gat_b[l]   = (const float*)d_in[base + 4];
    gn_w[l]    = (const float*)d_in[base + 5];
    gn_b[l]    = (const float*)d_in[base + 6];
    gn_ms[l]   = (const float*)d_in[base + 7];
    gru_wih[l] = (const float*)d_in[base + 8];
    gru_whh[l] = (const float*)d_in[base + 9];
    gru_bih[l] = (const float*)d_in[base + 10];
    gru_bhh[l] = (const float*)d_in[base + 11];
  }
  const float* mha_in_w  = (const float*)d_in[34]; const float* mha_in_b  = (const float*)d_in[35];
  const float* mha_out_w = (const float*)d_in[36]; const float* mha_out_b = (const float*)d_in[37];
  const float* tn_g = (const float*)d_in[38]; const float* tn_b = (const float*)d_in[39];
  const float* hw1[5]; const float* hb1[5]; const float* hw2[5]; const float* hb2[5];
  for (int k = 0; k < 5; ++k) {
    hw1[k] = (const float*)d_in[40 + 4 * k]; hb1[k] = (const float*)d_in[41 + 4 * k];
    hw2[k] = (const float*)d_in[42 + 4 * k]; hb2[k] = (const float*)d_in[43 + 4 * k];
  }
  char* ws = (char*)d_ws;
  float* out = (float*)d_out;
  auto F32 = [&](size_t o) { return (float*)(ws + o); };
  auto F16 = [&](size_t o) { return (_Float16*)(ws + o); };
  auto U32 = [&](size_t o) { return (unsigned*)(ws + o); };
  auto I32 = [&](size_t o) { return (int*)(ws + o); };
  auto GRID = [](long long n, int b) { return dim3((unsigned)((n + b - 1) / b)); };
  auto gemm = [&](const _Float16* A, const _Float16* BT, const float* bias,
                  float* C, _Float16* Ch, int M, int Nn, int K, int act) {
    dim3 g((unsigned)((M + 63) / 64), (unsigned)((Nn + 16 * GNT - 1) / (16 * GNT)));
    k_gemm_wmma<<<g, 128, 0, stream>>>(A, BT, bias, C, Ch, M, Nn, K, act);
  };
  auto zero32 = [&](float* p, long long n) { k_zero_f32<<<GRID(n, 256), 256, 0, stream>>>(p, n); };
  auto zero16 = [&](_Float16* p, long long n) { k_zero_f16<<<GRID(n, 256), 256, 0, stream>>>(p, n); };
  auto xpose = [&](const float* s, _Float16* d, int Ks, int Ns, int Kd) {
    long long n = (long long)Ns * Kd;
    k_transpose_f16<<<GRID(n, 256), 256, 0, stream>>>(s, d, Ks, Ns, Kd);
  };
  auto conv = [&](const float* s, _Float16* d, int rows, int cs, int cd) {
    long long n = (long long)rows * cd;
    k_convert_f16<<<GRID(n, 256), 256, 0, stream>>>(s, d, rows, cs, cd);
  };

  // ---- weight prep (f16, B pre-transposed to [N,K]) ----
  xpose(fe_w, F16(OFF_WFE), 72, HDIM, KCOMB);
  const size_t wgl[2] = {OFF_WGL0, OFF_WGL1}, wgr[2] = {OFF_WGR0, OFF_WGR1};
  const size_t wge[2] = {OFF_WGE0, OFF_WGE1};
  const size_t wih[2] = {OFF_WIH0, OFF_WIH1}, whh[2] = {OFF_WHH0, OFF_WHH1};
  for (int l = 0; l < 2; ++l) {
    xpose(gat_wl[l], F16(wgl[l]), HDIM, HDIM, HDIM);
    xpose(gat_wr[l], F16(wgr[l]), HDIM, HDIM, HDIM);
    xpose(gat_we[l], F16(wge[l]), HDIM, HDIM, HDIM);
    conv(gru_wih[l], F16(wih[l]), 3 * HDIM, HDIM, HDIM);  // x@W.T -> use W as BT
    conv(gru_whh[l], F16(whh[l]), 3 * HDIM, HDIM, HDIM);
  }
  conv(mha_in_w, F16(OFF_WMIN), 3 * HDIM, HDIM, HDIM);
  xpose(mha_out_w, F16(OFF_WMOUT), HDIM, HDIM, HDIM);
  xpose(hw1[0], F16(OFF_WOH1), HDIM, HDIM, HDIM);
  xpose(hw1[1], F16(OFF_WDH1), HDIM, HDIM, HDIM);
  xpose(hw1[2], F16(OFF_WEH1), HDIM, 64, HDIM);
  xpose(hw1[3], F16(OFF_WPH1), HDIM, HDIM, HDIM);
  xpose(hw1[4], F16(OFF_WCH1), HDIM, 32, HDIM);
  xpose(hw2[0], F16(OFF_WOH2), HDIM, 1, HDIM);
  xpose(hw2[1], F16(OFF_WDH2), HDIM, 4, HDIM);
  xpose(hw2[2], F16(OFF_WEH2), 64, 3, 64);
  xpose(hw2[3], F16(OFF_WPH2), HDIM, 4, HDIM);
  xpose(hw2[4], F16(OFF_WCH2), 32, 1, 32);

  // ---- graph prep: edge encoder, self-loop attrs, eproj ----
  const int* e_src = eidx;
  const int* e_dst = eidx + NEDGES;
  zero32(F32(A_LOOPS), (long long)NNODES * HDIM);
  zero32(F32(A_CNT), NNODES);
  k_count<<<GRID(NEDGES, 256), 256, 0, stream>>>(e_dst, F32(A_CNT), NEDGES);
  k_edge_enc<<<GRID((long long)NEDGES * HDIM, 256), 256, 0, stream>>>(
      e_attr, e_dst, ee_w, ee_b, F16(A_EATTR), F32(A_LOOPS));
  k_loopfill<<<GRID((long long)NNODES * HDIM, 256), 256, 0, stream>>>(
      F32(A_LOOPS), F32(A_CNT), F16(A_EATTR));
  k_fill_idx<<<GRID(ENF, 256), 256, 0, stream>>>(e_src, e_dst, I32(OFF_SRCF), I32(OFF_DSTF));
  const size_t eproj[2] = {A_EPROJ0, A_EPROJ1};
  for (int l = 0; l < 2; ++l)
    gemm(F16(A_EATTR), F16(wge[l]), nullptr, nullptr, F16(eproj[l]), ENF, HDIM, HDIM, 0);

  // ---- timestep loop: feature encoder + 2x (GATv2 + GraphNorm + residual) ----
  for (int t = 0; t < TSTEPS; ++t) {
    k_xcomb<<<GRID((long long)NNODES * KCOMB, 256), 256, 0, stream>>>(x_temporal, s_emb,
                                                                      F16(A_XCOMB), t);
    gemm(F16(A_XCOMB), F16(OFF_WFE), fe_b, F32(A_FEOUT), nullptr, NNODES, HDIM, KCOMB, 0);
    k_ln_relu<<<NNODES, 128, 0, stream>>>(F32(A_FEOUT), fe_lg, fe_lb, F32(A_H32), F16(A_H16), 1);
    for (int l = 0; l < 2; ++l) {
      gemm(F16(A_H16), F16(wgl[l]), nullptr, F32(A_XL), nullptr, NNODES, HDIM, HDIM, 0);
      gemm(F16(A_H16), F16(wgr[l]), nullptr, F32(A_XR), nullptr, NNODES, HDIM, HDIM, 0);
      k_fill_u32<<<GRID((long long)NNODES * NHEADS, 256), 256, 0, stream>>>(
          U32(A_SEGMAX), 0u, (long long)NNODES * NHEADS);
      zero32(F32(A_SEGDEN), (long long)NNODES * NHEADS);
      k_gat_logits<<<GRID((long long)ENF * NHEADS, 256), 256, 0, stream>>>(
          F32(A_XL), F32(A_XR), F16(eproj[l]), I32(OFF_SRCF), I32(OFF_DSTF),
          gat_att[l], F32(A_LOGITS), U32(A_SEGMAX));
      k_gat_exp<<<GRID((long long)ENF * NHEADS, 256), 256, 0, stream>>>(
          F32(A_LOGITS), U32(A_SEGMAX), F32(A_SEGDEN), I32(OFF_DSTF));
      k_bias_init<<<GRID((long long)NNODES * HDIM, 256), 256, 0, stream>>>(F32(A_GATOUT), gat_b[l]);
      k_gat_scatter<<<GRID((long long)ENF * HDIM, 256), 256, 0, stream>>>(
          F32(A_XL), F32(A_LOGITS), F32(A_SEGDEN), I32(OFF_SRCF), I32(OFF_DSTF), F32(A_GATOUT));
      k_colmean<<<HDIM, 256, 0, stream>>>(F32(A_GATOUT), F32(A_GNM));
      k_colvar<<<HDIM, 256, 0, stream>>>(F32(A_GATOUT), F32(A_GNM), gn_ms[l], F32(A_GNV));
      _Float16* h16out = (l == 1) ? (F16(OFF_SEQ16) + (size_t)t * NNODES * HDIM) : F16(A_H16);
      k_gn_apply<<<GRID((long long)NNODES * HDIM, 256), 256, 0, stream>>>(
          F32(A_GATOUT), F32(A_GNM), F32(A_GNV), gn_ms[l], gn_w[l], gn_b[l], F32(A_H32), h16out);
    }
  }

  // ---- 2-layer GRU over time (phase B: arena re-used) ----
  for (int l = 0; l < 2; ++l) {
    zero32(F32(B_GRUH32), (long long)NNODES * HDIM);
    zero16(F16(B_GRUH16), (long long)NNODES * HDIM);
    for (int t = 0; t < TSTEPS; ++t) {
      const _Float16* xin = (l == 0) ? (F16(OFF_SEQ16) + (size_t)t * NNODES * HDIM)
                                     : (F16(B_YS1) + (size_t)t * NNODES * HDIM);
      gemm(xin, F16(wih[l]), gru_bih[l], F32(B_GI), nullptr, NNODES, 3 * HDIM, HDIM, 0);
      gemm(F16(B_GRUH16), F16(whh[l]), gru_bhh[l], F32(B_GH), nullptr, NNODES, 3 * HDIM, HDIM, 0);
      _Float16* ys = (l == 0) ? (F16(B_YS1) + (size_t)t * NNODES * HDIM) : F16(B_HGRU16);
      float* ylast = (l == 1 && t == TSTEPS - 1) ? F32(B_HLAST) : nullptr;
      k_gru_pw<<<GRID((long long)NNODES * HDIM, 256), 256, 0, stream>>>(
          F32(B_GI), F32(B_GH), F32(B_GRUH32), F16(B_GRUH16), ys, ylast, t, l == 1 ? 1 : 0);
    }
  }

  // ---- MHA over time window ----
  gemm(F16(B_HGRU16), F16(OFF_WMIN), mha_in_b, nullptr, F16(B_QKV16),
       NNODES * TSTEPS, 3 * HDIM, HDIM, 0);
  k_attention<<<GRID((long long)NNODES * NHEADS, 256), 256, 0, stream>>>(
      F16(B_QKV16), F32(B_AW), F16(B_ATTN16));
  k_awmean<<<GRID((long long)NNODES * 36, 256), 256, 0, stream>>>(F32(B_AW), out + O_AW);
  gemm(F16(B_ATTN16), F16(OFF_WMOUT), mha_out_b, F32(B_HATTN), nullptr, NNODES, HDIM, HDIM, 0);
  k_addln<<<NNODES, 128, 0, stream>>>(F32(B_HLAST), F32(B_HATTN), tn_g, tn_b,
                                      F32(B_HFIN32), F16(B_HFIN16));

  // ---- output heads ----
  // order = relu(relu(x@w1+b1)@w2+b2)
  gemm(F16(B_HFIN16), F16(OFF_WOH1), hb1[0], nullptr, F16(B_MID16), NNODES, HDIM, HDIM, 1);
  gemm(F16(B_MID16), F16(OFF_WOH2), hb2[0], out + O_ORDER, nullptr, NNODES, 1, HDIM, 1);
  // demand (linear)
  gemm(F16(B_HFIN16), F16(OFF_WDH1), hb1[1], nullptr, F16(B_MID16), NNODES, HDIM, HDIM, 1);
  gemm(F16(B_MID16), F16(OFF_WDH2), hb2[1], out + O_DEM, nullptr, NNODES, 4, HDIM, 0);
  // exc (softmax over 3)
  gemm(F16(B_HFIN16), F16(OFF_WEH1), hb1[2], nullptr, F16(B_MID16), NNODES, 64, HDIM, 1);
  gemm(F16(B_MID16), F16(OFF_WEH2), hb2[2], F32(B_EXCT), nullptr, NNODES, 3, 64, 0);
  k_softmax3<<<GRID(NNODES, 256), 256, 0, stream>>>(F32(B_EXCT), out + O_EXC);
  // prop (sigmoid)
  gemm(F16(B_HFIN16), F16(OFF_WPH1), hb1[3], nullptr, F16(B_MID16), NNODES, HDIM, HDIM, 1);
  gemm(F16(B_MID16), F16(OFF_WPH2), hb2[3], out + O_PROP, nullptr, NNODES, 4, HDIM, 2);
  // conf (sigmoid)
  gemm(F16(B_HFIN16), F16(OFF_WCH1), hb1[4], nullptr, F16(B_MID16), NNODES, 32, HDIM, 1);
  gemm(F16(B_MID16), F16(OFF_WCH2), hb2[4], out + O_CONF, nullptr, NNODES, 1, 32, 2);
}
